// RelationalNeigborConstrastLoss_11501922418701
// MI455X (gfx1250) — compile-verified, matrix operand order checked by reading, not run on hardware
//
#include <hip/hip_runtime.h>
#include <stdint.h>

// ---------------------------------------------------------------------------
// Problem constants (match reference)
// ---------------------------------------------------------------------------
#define DIMS   128          // D
#define NPAIR  2048         // P
#define NSPLIT 16           // group-range splits per p-tile for parallelism

typedef _Float16 h8   __attribute__((ext_vector_type(8)));
typedef _Float16 v16h __attribute__((ext_vector_type(16)));
typedef float    v8f  __attribute__((ext_vector_type(8)));

// ---------------------------------------------------------------------------
// Kernel 1a: f32 -> f16 conversion of an encoding table
// ---------------------------------------------------------------------------
__global__ void cvt_f32_f16(const float* __restrict__ src,
                            _Float16* __restrict__ dst, int n) {
  int i = blockIdx.x * blockDim.x + threadIdx.x;
  if (i < n) dst[i] = (_Float16)src[i];
}

// ---------------------------------------------------------------------------
// Kernel 1b: gather rows enc[idx[p]] -> out[p]  (P*DIMS threads)
// ---------------------------------------------------------------------------
__global__ void gather_rows(const _Float16* __restrict__ enc,
                            const int* __restrict__ idx,
                            _Float16* __restrict__ out) {
  int t = blockIdx.x * blockDim.x + threadIdx.x;   // P*DIMS total
  int p = t >> 7;          // /128
  int k = t & (DIMS - 1);
  out[t] = enc[(size_t)idx[p] * DIMS + k];
}

// ---------------------------------------------------------------------------
// Kernel 2: gather + pack adjacency mask, transposed:
//   maskT[n][p/32] bit (p%32) = (adj[n, idx[p]] != 0)
// One wave handles 32 consecutive p for one n; ballot packs a u32 word.
// This pass carries the unavoidable ~400MB/side scattered adjacency traffic.
// ---------------------------------------------------------------------------
__global__ void build_mask(const int* __restrict__ adj,
                           const int* __restrict__ idx,
                           uint32_t* __restrict__ maskT, int No) {
  int t    = blockIdx.x * blockDim.x + threadIdx.x;
  int lane = t & 31;
  int w    = t >> 5;
  int pw   = w & (NPAIR / 32 - 1);   // word within row (64 words)
  int n    = w >> 6;
  if (n >= No) return;               // grid sized exactly; waves stay whole
  int col = idx[(pw << 5) + lane];
  int v   = adj[(size_t)n * No + col];
  unsigned long long b = __ballot(v != 0);
  if (lane == 0) maskT[(size_t)n * (NPAIR / 32) + pw] = (uint32_t)b;
}

// ---------------------------------------------------------------------------
// Kernel 2b: repack mask into the main-loop consumption layout:
//   maskPT[pt][g][lane] : u32 whose byte b is the mask byte for
//     chunk c = 4g+b, rows p = 16*pt + 8*(lane>>4) .. +7, column n = 16c+(lane&15)
// One coalesced 128B wave-load then serves 4 chunks; chunks >= nchunks are 0.
// ---------------------------------------------------------------------------
__global__ void repack_mask(const uint8_t* __restrict__ maskTb, // [No][P/8]
                            uint32_t* __restrict__ maskPT,      // [P/16][ngroup][32]
                            int nchunks, int ngroup) {
  int tid   = blockIdx.x * blockDim.x + threadIdx.x;
  int total = (NPAIR / 16) * ngroup * 32;
  if (tid >= total) return;
  int lane = tid & 31;
  int g    = (tid >> 5) % ngroup;
  int pt   = tid / (ngroup * 32);
  int colB = pt * 2 + (lane >> 4);
  int lq   = lane & 15;
  uint32_t w = 0;
#pragma unroll
  for (int b = 0; b < 4; ++b) {
    int c = g * 4 + b;
    if (c < nchunks) {
      int n = (c << 4) + lq;
      w |= (uint32_t)maskTb[(size_t)n * (NPAIR / 8) + colB] << (8 * b);
    }
  }
  maskPT[tid] = w;
}

// ---------------------------------------------------------------------------
// 16-bit WMMA fragment load from a row-major row with contiguous K.
// CDNA5 16x16x32 f16 layout: lanes 0-15 hold K {0..7,16..23} of row/col L,
// lanes 16-31 hold K {8..15,24..31} of row/col L-16 (A and B symmetric).
// -> two b128 loads per lane.
// ---------------------------------------------------------------------------
__device__ __forceinline__ v16h load_frag(const _Float16* __restrict__ row,
                                          int kbase, int hi) {
  const int o1 = hi ? 8 : 0;
  const int o2 = hi ? 24 : 16;
  h8 a = *(const h8*)(row + kbase + o1);
  h8 b = *(const h8*)(row + kbase + o2);
  v16h r;
#pragma unroll
  for (int i = 0; i < 8; ++i) { r[i] = a[i]; r[i + 8] = b[i]; }
  return r;
}

// ---------------------------------------------------------------------------
// Kernel 3: flash-style masked online softmax over key rows via WMMA.
// grid = (NPAIR/16, NSPLIT), block = 1 wave (32 threads).
// Mask words are double-buffered one group (4 chunks) ahead so the skip
// decision never stalls on a fresh load.
// ---------------------------------------------------------------------------
__global__ __launch_bounds__(32)
void npairs_partial(const _Float16* __restrict__ anchorG,   // [P,DIMS]
                    const _Float16* __restrict__ queryG,    // [P,DIMS]
                    const _Float16* __restrict__ keys,      // [No,DIMS]
                    const uint32_t* __restrict__ maskPT,    // [P/16][ngroup][32]
                    int ngroup,
                    float4* __restrict__ partials) {        // [NSPLIT][P]
  const int lane  = threadIdx.x;
  const int hi    = lane >> 4;     // which 16-lane half
  const int lq    = lane & 15;     // row/col within tile
  const int pt    = blockIdx.x;    // p tile (16 pairs)
  const int split = blockIdx.y;
  const int pbase = pt * 16;

  const int per = (ngroup + NSPLIT - 1) / NSPLIT;
  const int g0  = split * per;
  const int g1  = (g0 + per < ngroup) ? (g0 + per) : ngroup;

  // Persistent A-operand fragments: query rows (for S) and anchor rows (pos).
  const _Float16* qrow = queryG  + (size_t)(pbase + lq) * DIMS;
  const _Float16* arow = anchorG + (size_t)(pbase + lq) * DIMS;
  v16h QF[4], AF[4];
#pragma unroll
  for (int kc = 0; kc < 4; ++kc) {
    QF[kc] = load_frag(qrow, kc * 32, hi);
    AF[kc] = load_frag(arow, kc * 32, hi);
  }

  // Online-softmax state per accumulator slot v (row p = pbase + v + 8*hi).
  float m[8], d[8], num[8], cnt[8];
#pragma unroll
  for (int v = 0; v < 8; ++v) { m[v] = -3.0e38f; d[v] = 0.f; num[v] = 0.f; cnt[v] = 0.f; }

  const uint32_t* mbase = maskPT + (size_t)pt * ngroup * 32 + lane;
  uint32_t mw = (g0 < g1) ? mbase[(size_t)g0 * 32] : 0u;

  for (int g = g0; g < g1; ++g) {
    // prefetch next group's mask word while this group computes
    uint32_t mwn = (g + 1 < g1) ? mbase[(size_t)(g + 1) * 32] : 0u;
#pragma unroll
    for (int b = 0; b < 4; ++b) {
      unsigned mb = (mw >> (8 * b)) & 0xffu;
      if (__ballot(mb != 0) == 0ull) continue;   // empty 16x16 tile: skip all

      const int n = (((g << 2) + b) << 4) + lq;
      const _Float16* krow = keys + (size_t)n * DIMS;
      v8f S, PP;
#pragma unroll
      for (int i = 0; i < 8; ++i) { S[i] = 0.f; PP[i] = 0.f; }
#pragma unroll
      for (int kc = 0; kc < 4; ++kc) {
        v16h B = load_frag(krow, kc * 32, hi);
        S  = __builtin_amdgcn_wmma_f32_16x16x32_f16(false, QF[kc], false, B,
                                                    (short)0, S,  false, false);
        PP = __builtin_amdgcn_wmma_f32_16x16x32_f16(false, AF[kc], false, B,
                                                    (short)0, PP, false, false);
      }
#pragma unroll
      for (int v = 0; v < 8; ++v) {
        if ((mb >> v) & 1u) {
          float s  = S[v];
          float mn = fmaxf(m[v], s);
          float sc = __expf(m[v] - mn);   // rescale old terms
          float es = __expf(s - mn);
          float t  = PP[v] - 1.0f;
          d[v]   = d[v]   * sc + es;
          num[v] = num[v] * sc + es * t * t;
          m[v]   = mn;
          cnt[v] += 1.0f;
        }
      }
    }
    mw = mwn;
  }

  // Merge softmax states across the 16 lanes of each half-wave.
#pragma unroll
  for (int off = 1; off <= 8; off <<= 1) {
#pragma unroll
    for (int v = 0; v < 8; ++v) {
      float m2 = __shfl_xor(m[v],   off, 32);
      float d2 = __shfl_xor(d[v],   off, 32);
      float n2 = __shfl_xor(num[v], off, 32);
      float c2 = __shfl_xor(cnt[v], off, 32);
      float mn = fmaxf(m[v], m2);
      float s1 = __expf(m[v] - mn);
      float s2 = __expf(m2   - mn);
      d[v]   = d[v]   * s1 + d2 * s2;
      num[v] = num[v] * s1 + n2 * s2;
      m[v]   = mn;
      cnt[v] += c2;
    }
  }

  if (lq == 0) {
#pragma unroll
    for (int v = 0; v < 8; ++v) {
      int p = pbase + v + hi * 8;
      partials[(size_t)split * NPAIR + p] = make_float4(m[v], d[v], num[v], cnt[v]);
    }
  }
}

// ---------------------------------------------------------------------------
// Kernel 4: merge splits, x[p] = K * num/d, loss = sum(x)/(1+#nonempty), both
// sides summed. Single block -> deterministic.
// ---------------------------------------------------------------------------
__global__ __launch_bounds__(256)
void finalize(const float4* __restrict__ partA,
              const float4* __restrict__ partB,
              float* __restrict__ out) {
  __shared__ float ssum[256];
  __shared__ float scnt[256];
  const int tid = threadIdx.x;
  float loss = 0.f;
  for (int side = 0; side < 2; ++side) {
    const float4* pp = side ? partB : partA;
    float lsum = 0.f, lcnt = 0.f;
    for (int p = tid; p < NPAIR; p += 256) {
      float m = -3.0e38f, d = 0.f, num = 0.f, c = 0.f;
      for (int s = 0; s < NSPLIT; ++s) {
        float4 st = pp[(size_t)s * NPAIR + p];
        float mn = fmaxf(m, st.x);
        float s1 = __expf(m - mn);
        float s2 = __expf(st.x - mn);
        d   = d   * s1 + st.y * s2;
        num = num * s1 + st.z * s2;
        m   = mn;
        c  += st.w;
      }
      if (c > 0.f) { lsum += c * num / d; lcnt += 1.f; }
    }
    ssum[tid] = lsum; scnt[tid] = lcnt;
    __syncthreads();
    for (int o = 128; o > 0; o >>= 1) {
      if (tid < o) { ssum[tid] += ssum[tid + o]; scnt[tid] += scnt[tid + o]; }
      __syncthreads();
    }
    if (tid == 0) loss += ssum[0] / (1.0f + scnt[0]);
    __syncthreads();
  }
  if (tid == 0) out[0] = loss;
}

// ---------------------------------------------------------------------------
// Host launcher
// ---------------------------------------------------------------------------
extern "C" void kernel_launch(void* const* d_in, const int* in_sizes, int n_in,
                              void* d_out, int out_size, void* d_ws, size_t ws_size,
                              hipStream_t stream) {
  const int*   typeA_idx = (const int*)d_in[0];
  const int*   typeB_idx = (const int*)d_in[1];
  const float* encA      = (const float*)d_in[2];
  const float* encB      = (const float*)d_in[3];
  const int*   adjA      = (const int*)d_in[4];
  const int*   adjB      = (const int*)d_in[5];

  const int NA = in_sizes[2] / DIMS;   // 10000
  const int NB = in_sizes[3] / DIMS;   // 10000

  const int chunksA = NB / 16;                  // side A iterates NB keys
  const int chunksB = NA / 16;                  // side B iterates NA keys
  const int ngroupA = (chunksA + 3) / 4;
  const int ngroupB = (chunksB + 3) / 4;

  // Workspace carve-up (~17.6 MB total)
  char* ws = (char*)d_ws;
  size_t off = 0;
  auto carve = [&](size_t bytes) -> void* {
    void* p = ws + off;
    off += (bytes + 255) & ~(size_t)255;
    return p;
  };
  _Float16* encA_h  = (_Float16*)carve((size_t)NA * DIMS * sizeof(_Float16));
  _Float16* encB_h  = (_Float16*)carve((size_t)NB * DIMS * sizeof(_Float16));
  _Float16* GA      = (_Float16*)carve((size_t)NPAIR * DIMS * sizeof(_Float16)); // encA[typeA_idx]
  _Float16* GB      = (_Float16*)carve((size_t)NPAIR * DIMS * sizeof(_Float16)); // encB[typeB_idx]
  uint32_t* maskA   = (uint32_t*)carve((size_t)NB * (NPAIR / 32) * 4);  // side A: typeB_adj/typeB_idx
  uint32_t* maskB   = (uint32_t*)carve((size_t)NA * (NPAIR / 32) * 4);  // side B: typeA_adj/typeA_idx
  uint32_t* maskPTA = (uint32_t*)carve((size_t)(NPAIR / 16) * ngroupA * 32 * 4);
  uint32_t* maskPTB = (uint32_t*)carve((size_t)(NPAIR / 16) * ngroupB * 32 * 4);
  float4*   partA   = (float4*)carve((size_t)NSPLIT * NPAIR * sizeof(float4));
  float4*   partB   = (float4*)carve((size_t)NSPLIT * NPAIR * sizeof(float4));
  (void)ws_size; (void)n_in; (void)out_size;

  // 1) convert encodings to f16
  {
    int n = NA * DIMS;
    cvt_f32_f16<<<(n + 255) / 256, 256, 0, stream>>>(encA, encA_h, n);
    n = NB * DIMS;
    cvt_f32_f16<<<(n + 255) / 256, 256, 0, stream>>>(encB, encB_h, n);
  }
  // 1b) gather anchor/query rows
  {
    int n = NPAIR * DIMS;
    gather_rows<<<n / 256, 256, 0, stream>>>(encA_h, typeA_idx, GA);
    gather_rows<<<n / 256, 256, 0, stream>>>(encB_h, typeB_idx, GB);
  }
  // 2) build transposed bitmasks (the 800MB adjacency gather lives here)
  {
    int threads = NB * (NPAIR / 32) * 32;
    build_mask<<<threads / 256, 256, 0, stream>>>(adjB, typeB_idx, maskA, NB);
    threads = NA * (NPAIR / 32) * 32;
    build_mask<<<threads / 256, 256, 0, stream>>>(adjA, typeA_idx, maskB, NA);
  }
  // 2b) repack masks into coalesced per-p-tile group layout
  {
    int total = (NPAIR / 16) * ngroupA * 32;
    repack_mask<<<(total + 255) / 256, 256, 0, stream>>>(
        (const uint8_t*)maskA, maskPTA, chunksA, ngroupA);
    total = (NPAIR / 16) * ngroupB * 32;
    repack_mask<<<(total + 255) / 256, 256, 0, stream>>>(
        (const uint8_t*)maskB, maskPTB, chunksB, ngroupB);
  }
  // 3) WMMA flash-softmax partials, one wave per (p-tile, split)
  {
    dim3 grid(NPAIR / 16, NSPLIT);
    // side A: anchor=GA, query=GB, keys=encB, mask from typeB_adj
    npairs_partial<<<grid, 32, 0, stream>>>(GA, GB, encB_h, maskPTA, ngroupA, partA);
    // side B: anchor=GB, query=GA, keys=encA, mask from typeA_adj
    npairs_partial<<<grid, 32, 0, stream>>>(GB, GA, encA_h, maskPTB, ngroupB, partB);
  }
  // 4) reduce to scalar
  finalize<<<1, 256, 0, stream>>>(partA, partB, (float*)d_out);
}
